// ISTFT_75720273428951
// MI455X (gfx1250) — matrix operand order repaired and checked by reading, not compiled, not run on hardware
//
#include <hip/hip_runtime.h>

// ---------------- problem constants ----------------
#define BATCH     16
#define NFREQ     513
#define TFRM      2048
#define WINL      1024
#define HOP       256
#define NFFT      1024
#define PADT      384
#define OUT_PER_B 524288            // TFRM * HOP  (trimmed length per batch)
#define ENV_LEN   525056            // (TFRM-1)*HOP + WINL
#define NT_TILES  65                // ceil(2051 t0-positions / 32)
#define KSEG      17                // ceil(513 / 32)
#define KB_TOTAL  136               // 2 parts * 4 shifts * 17 k-blocks
#define A_ELEMS   (KB_TOTAL * 8192) // per kb: 256 rows * 32 k  (bf16)
#define ENV_BYTES (ENV_LEN * 4)

#define SA_BYTES  (256 * 48 * 2)    // 24576 : A buffer, row stride 48 bf16 (96 B)
#define SB_BYTES  (32 * 48 * 2)     // 3072  : B buffer
#define SMEM_BYTES (2 * SA_BYTES + 2 * SB_BYTES)  // 55296

typedef __attribute__((ext_vector_type(16))) __bf16       v16bf;
typedef __attribute__((ext_vector_type(8)))  __bf16       v8bf;
typedef __attribute__((ext_vector_type(8)))  float        v8f;
typedef __attribute__((ext_vector_type(4)))  unsigned int v4u;
typedef __attribute__((ext_vector_type(8)))  int          v8i;
typedef __attribute__((ext_vector_type(4)))  int          v4i;

__device__ __forceinline__ unsigned short f2bf_bits(float f) {
    __bf16 h = (__bf16)f;
    return __builtin_bit_cast(unsigned short, h);
}

// ---------------- kernel 1: window^2 overlap-add envelope ----------------
__global__ void istft_env_kernel(float* __restrict__ env) {
    int p = blockIdx.x * blockDim.x + threadIdx.x;
    if (p >= ENV_LEN) return;
    const float c2pi = 6.28318530717958647692f / (float)WINL;
    int tlo = (p >= WINL - 1) ? ((p - (WINL - 1) + HOP - 1) / HOP) : 0;
    int thi = p / HOP; if (thi > TFRM - 1) thi = TFRM - 1;
    float s = 0.f;
    for (int t = tlo; t <= thi; ++t) {
        int n = p - t * HOP;
        float w = 0.5f * (1.0f - cosf((float)n * c2pi));
        s += w * w;
    }
    env[p] = s;
}

// ---------------- kernel 2: fused window * iDFT coefficients (bf16) ----------------
__global__ void istft_coef_kernel(unsigned short* __restrict__ Aws) {
    int idx = blockIdx.x * blockDim.x + threadIdx.x;
    if (idx >= A_ELEMS) return;
    int kb   = idx >> 13;
    int rem  = idx & 8191;
    int r    = rem >> 5;
    int kk   = rem & 31;
    int part = kb / (4 * KSEG);
    int rem2 = kb % (4 * KSEG);
    int j    = rem2 / KSEG;
    int k    = (rem2 % KSEG) * 32 + kk;
    int n    = r + (j << 8);
    float v = 0.f;
    if (k < NFREQ) {
        const float c2pi = 6.28318530717958647692f / (float)NFFT;
        float wn = 0.5f * (1.0f - cosf((float)n * c2pi));   // periodic Hann
        int   m  = (k * n) & (NFFT - 1);                    // exact period reduction
        float ang = (float)m * c2pi;
        bool edge = (k == 0) || (k == NFFT / 2);
        float ck  = edge ? 1.0f : 2.0f;
        if (part == 0)       v = ck * (1.0f / (float)NFFT) * cosf(ang) * wn;
        else if (!edge)      v = -ck * (1.0f / (float)NFFT) * sinf(ang) * wn;
    }
    Aws[idx] = f2bf_bits(v);
}

// ---------------- TDM: DMA one contiguous 16 KB A tile into LDS ----------------
// D# describes a 1-row tensor of 4096 dwords with LDS padding of 8 dwords every
// 16 dwords => 64B data + 32B pad = the stride-48-bf16 LDS row layout.
__device__ __forceinline__ void tdm_load_A(const __bf16* gsrc, __bf16* ldst) {
    unsigned long long ga = (unsigned long long)(uintptr_t)gsrc;
    unsigned           la = (unsigned)(uintptr_t)ldst;
    v4u g0;
    g0.x = 1u;                                             // count=1 (valid user D#)
    g0.y = la;                                             // lds_addr
    g0.z = (unsigned)ga;                                   // global_addr[31:0]
    g0.w = (unsigned)((ga >> 32) & 0x01FFFFFFull) | (2u << 30); // addr[56:32], type=2
    v8i g1;
    g1[0] = (int)((2u << 16)        // data_size = 4 bytes
                | (1u << 20)        // pad_enable
                | (3u << 22)        // pad_interval: every 16 dwords (64 B)
                | (7u << 25));      // pad_amount: 8 dwords (32 B)
    g1[1] = (int)(0x1000u << 16);   // tensor_dim0 = 4096 (low 16 bits)
    g1[2] = (int)(1u << 16);        // tensor_dim0 hi = 0, tensor_dim1 = 1 (low 16)
    g1[3] = (int)(0x1000u << 16);   // tensor_dim1 hi = 0, tile_dim0 = 4096
    g1[4] = 1;                      // tile_dim1 = 1, tile_dim2 = 0
    g1[5] = 4096;                   // tensor_dim0_stride low 32
    g1[6] = 0;
    g1[7] = 0;
    v4i gz = {0, 0, 0, 0};
#if __clang_major__ >= 23
    v8i gz8 = {0, 0, 0, 0, 0, 0, 0, 0};
    __builtin_amdgcn_tensor_load_to_lds(g0, g1, gz, gz, gz8, 0);
#else
    __builtin_amdgcn_tensor_load_to_lds(g0, g1, gz, gz, 0);
#endif
}

// ---------------- B tile staging: gather + f32->bf16 + transpose ----------------
__device__ __forceinline__ void stage_B(const float* __restrict__ spec_re,
                                        const float* __restrict__ spec_im,
                                        int b, int t0base, int kb,
                                        int kkB, int nn0, __bf16* __restrict__ sBbuf) {
    const int part = kb / (4 * KSEG);
    const int rem2 = kb % (4 * KSEG);
    const int j    = rem2 / KSEG;
    const int k    = (rem2 % KSEG) * 32 + kkB;
    const float* srow = (part ? spec_im : spec_re)
                      + ((size_t)(b * NFREQ + (k < NFREQ ? k : 0))) * TFRM;
    #pragma unroll
    for (int u = 0; u < 4; ++u) {
        int t = t0base + nn0 + u - j;
        float f = 0.f;
        if (k < NFREQ && t >= 0 && t < TFRM) f = srow[t];
        sBbuf[(nn0 + u) * 48 + kkB] = (__bf16)f;
    }
}

// ---------------- kernel 3: fused GEMM(+OLA), TDM + double buffer + WMMA ----------------
__global__ __launch_bounds__(256) void istft_wmma_kernel(
    const float*          __restrict__ spec_re,
    const float*          __restrict__ spec_im,
    const unsigned short* __restrict__ AwsU,
    const float*          __restrict__ env,
    float*                __restrict__ out)
{
    __shared__ __align__(16) char smem[SMEM_BYTES];
    // buffer pointers are computed from offsets at each use site (no pointer
    // arrays: addrspacecast of LDS pointers must not become static initializers)

    const __bf16* Aws = (const __bf16*)AwsU;
    const int tid    = threadIdx.x;
    const int b      = blockIdx.x / NT_TILES;
    const int t0base = (blockIdx.x % NT_TILES) * 32;

    const int  lane  = tid & 31;
    const int  lr    = lane & 15;
    const bool hi    = lane >= 16;
    const int  Mbase = (tid >> 5) << 5;

    // scalar (SCC) guard: only wave 0 issues TDM ops (TDM ignores EXEC)
    const bool isWave0 = (__builtin_amdgcn_readfirstlane((int)tid) < 32);

    const int kkB = tid >> 3;         // 0..31 : k within 32-block
    const int nn0 = (tid & 7) << 2;   // 0,4,..,28 : 4 consecutive columns

    __builtin_prefetch(env + (size_t)t0base * HOP + tid, 0, 1);

    v8f acc00 = {}, acc01 = {}, acc10 = {}, acc11 = {};

    // prologue: stage tile 0 into buffer 0
    if (isWave0) tdm_load_A(Aws, (__bf16*)smem);
    stage_B(spec_re, spec_im, b, t0base, 0, kkB, nn0,
            (__bf16*)(smem + 2 * SA_BYTES));
    __builtin_amdgcn_s_wait_tensorcnt(0);
    __syncthreads();

    for (int kb = 0; kb < KB_TOTAL; ++kb) {
        const int cur = kb & 1, nxt = cur ^ 1;

        // stage next tile while computing current one
        if (kb + 1 < KB_TOTAL) {
            if (isWave0)
                tdm_load_A(Aws + ((size_t)(kb + 1) << 13),
                           (__bf16*)(smem + nxt * SA_BYTES));
            stage_B(spec_re, spec_im, b, t0base, kb + 1, kkB, nn0,
                    (__bf16*)(smem + 2 * SA_BYTES + nxt * SB_BYTES));
        }

        // fragments per ISA 16-bit A(16x32) / B(32x16) lane layouts
        const __bf16* cA = (const __bf16*)(smem + cur * SA_BYTES);
        const __bf16* cB = (const __bf16*)(smem + 2 * SA_BYTES + cur * SB_BYTES);
        v16bf a0, a1, b0, b1;
        {
            const int c = hi ? 8 : 0;
            int m0 = Mbase + lr;
            v8bf lo = *(const v8bf*)(cA + m0 * 48 + c);
            v8bf hh = *(const v8bf*)(cA + m0 * 48 + c + 16);
            #pragma unroll
            for (int e = 0; e < 8; ++e) { a0[e] = lo[e]; a0[8 + e] = hh[e]; }
            int m1 = Mbase + 16 + lr;
            lo = *(const v8bf*)(cA + m1 * 48 + c);
            hh = *(const v8bf*)(cA + m1 * 48 + c + 16);
            #pragma unroll
            for (int e = 0; e < 8; ++e) { a1[e] = lo[e]; a1[8 + e] = hh[e]; }
        }
        {
            const int c = hi ? 8 : 0;
            int n0 = lr;
            v8bf lo = *(const v8bf*)(cB + n0 * 48 + c);
            v8bf hh = *(const v8bf*)(cB + n0 * 48 + c + 16);
            #pragma unroll
            for (int e = 0; e < 8; ++e) { b0[e] = lo[e]; b0[8 + e] = hh[e]; }
            int n1 = 16 + lr;
            lo = *(const v8bf*)(cB + n1 * 48 + c);
            hh = *(const v8bf*)(cB + n1 * 48 + c + 16);
            #pragma unroll
            for (int e = 0; e < 8; ++e) { b1[e] = lo[e]; b1[8 + e] = hh[e]; }
        }

        acc00 = __builtin_amdgcn_wmma_f32_16x16x32_bf16(false, a0, false, b0, (short)0, acc00, false, false);
        acc01 = __builtin_amdgcn_wmma_f32_16x16x32_bf16(false, a0, false, b1, (short)0, acc01, false, false);
        acc10 = __builtin_amdgcn_wmma_f32_16x16x32_bf16(false, a1, false, b0, (short)0, acc10, false, false);
        acc11 = __builtin_amdgcn_wmma_f32_16x16x32_bf16(false, a1, false, b1, (short)0, acc11, false, false);

        // next-tile staging must be complete before the buffers swap
        __builtin_amdgcn_s_wait_tensorcnt(0);
        __syncthreads();
    }

    // epilogue: accumulators -> LDS (stride 33 = conflict-free) -> coalesced store
    float* sC = (float*)smem;   // overlay, safe after final barrier
    #pragma unroll
    for (int e = 0; e < 8; ++e) {
        int M = e + (hi ? 8 : 0);
        sC[(Mbase +      M) * 33 + (     lr)] = acc00[e];
        sC[(Mbase +      M) * 33 + (16 + lr)] = acc01[e];
        sC[(Mbase + 16 + M) * 33 + (     lr)] = acc10[e];
        sC[(Mbase + 16 + M) * 33 + (16 + lr)] = acc11[e];
    }
    __syncthreads();

    #pragma unroll 4
    for (int n = 0; n < 32; ++n) {
        long p = (long)(t0base + n) * HOP + tid;
        long q = p - PADT;
        if (q >= 0 && q < OUT_PER_B) {
            out[(size_t)b * OUT_PER_B + q] = sC[tid * 33 + n] / env[p];
        }
    }
}

// ---------------- launcher ----------------
extern "C" void kernel_launch(void* const* d_in, const int* in_sizes, int n_in,
                              void* d_out, int out_size, void* d_ws, size_t ws_size,
                              hipStream_t stream) {
    (void)in_sizes; (void)n_in; (void)out_size; (void)ws_size;
    const float* spec_re = (const float*)d_in[0];
    const float* spec_im = (const float*)d_in[1];
    float* out = (float*)d_out;

    float*          env = (float*)d_ws;
    unsigned short* Aws = (unsigned short*)((char*)d_ws + ENV_BYTES);

    istft_env_kernel <<<(ENV_LEN + 255) / 256, 256, 0, stream>>>(env);
    istft_coef_kernel<<<(A_ELEMS + 255) / 256, 256, 0, stream>>>(Aws);
    istft_wmma_kernel<<<BATCH * NT_TILES, 256, 0, stream>>>(spec_re, spec_im, Aws, env, out);
}
